// GAT_based_on_dynamic_prior_knowledge_39084202394200
// MI455X (gfx1250) — compile-verified, matrix operand order checked by reading, not run on hardware
//
#include <hip/hip_runtime.h>
#include <hip/hip_bf16.h>

// GAT with dynamic prior knowledge — fully fused per-batch-sample kernel for
// gfx1250 (MI455X). One workgroup (256 threads = 8 wave32) per batch sample.
// All GEMMs via v_wmma_f32_16x16x32_bf16; softmax/blend in f32 VALU; all
// intermediates live in 305 KB of WGP LDS so the [B,N,N] attention tensors
// never touch HBM. Wh / Wh_o are stored TRANSPOSED in LDS so WMMA B-fragment
// K-pairs are contiguous (u32 DS loads) and accumulator scatter is one
// 16-byte DS store per lane. bf16 packing uses v_cvt_pk_bf16_f32.

#define NN    128   // nodes
#define NF    256   // input features
#define NHID  64
#define NHEAD 8
#define NCLS  64
#define ALPHA 0.2f
#define BETA  0.5f
#define NEGV  -9.0e15f

typedef __attribute__((ext_vector_type(16))) __bf16 v16bf;
typedef __attribute__((ext_vector_type(8)))  float  v8f;
typedef __attribute__((ext_vector_type(2)))  __bf16 bf16x2;

union FragU {
    v16bf v;
    unsigned u[8];
    unsigned short h[16];
};

#define DEVINL static __device__ __forceinline__

DEVINL unsigned pack_bf2(float a, float b) {   // one v_cvt_pk_bf16_f32
#if __has_builtin(__builtin_amdgcn_cvt_pk_bf16_f32)
    auto v = __builtin_amdgcn_cvt_pk_bf16_f32(a, b);
    return __builtin_bit_cast(unsigned, v);
#else
    bf16x2 v;
    v[0] = (__bf16)a;
    v[1] = (__bf16)b;
    return __builtin_bit_cast(unsigned, v);
#endif
}
DEVINL unsigned short f2bf_bits(float f) {
    __bf16 b = (__bf16)f;
    return __builtin_bit_cast(unsigned short, b);
}
DEVINL float bf2f(unsigned short h) {
    unsigned u = ((unsigned)h) << 16;
    return __builtin_bit_cast(float, u);
}

DEVINL v8f wmma_bf16(v16bf a, v16bf b, v8f c) {
    // (neg_a, A, neg_b, B, c_mod, C, reuse_a, reuse_b)
    return __builtin_amdgcn_wmma_f32_16x16x32_bf16(false, a, false, b,
                                                   (short)0, c, false, false);
}

// A-matrix 16x32 bf16 fragment from LDS, A row-major [M][K] stride ld.
// Lane L: M = L&15 ; VGPR v holds K pair at K = (v>=4?16:0) + (L>=16?8:0) + (v&3)*2.
DEVINL v16bf load_a_lds(const unsigned short* p, int row0, int col0, int ld, int lane) {
    const int m = lane & 15, hi = lane >> 4;
    FragU f;
#pragma unroll
    for (int v = 0; v < 8; ++v) {
        const int kb = ((v >= 4) ? 16 : 0) + hi * 8 + (v & 3) * 2;
        f.u[v] = *(const unsigned*)(p + (row0 + m) * ld + col0 + kb);
    }
    return f.v;
}

// B-matrix 32x16 bf16 fragment from LDS, B stored TRANSPOSED [N][K] stride ldT.
// K-pairs are contiguous -> u32 loads.
DEVINL v16bf load_b_ldsT(const unsigned short* p, int k0, int col0, int ldT, int lane) {
    const int n = lane & 15, hi = lane >> 4;
    FragU f;
#pragma unroll
    for (int v = 0; v < 8; ++v) {
        const int kb = ((v >= 4) ? 16 : 0) + hi * 8 + (v & 3) * 2;
        f.u[v] = *(const unsigned*)(p + (col0 + n) * ldT + k0 + kb);
    }
    return f.v;
}

// B-matrix 32x16 fragment from global f32 [K][N] stride ld, converted to bf16.
DEVINL v16bf load_b_g(const float* p, int k0, int col0, int ld, int lane) {
    const int n = lane & 15, hi = lane >> 4;
    FragU f;
#pragma unroll
    for (int v = 0; v < 8; ++v) {
        const int kb = ((v >= 4) ? 16 : 0) + hi * 8 + (v & 3) * 2;
        f.u[v] = pack_bf2(p[(k0 + kb) * ld + col0 + n],
                          p[(k0 + kb + 1) * ld + col0 + n]);
    }
    return f.v;
}

// Accumulator (16x16 f32 C/D layout) -> 8 contiguous bf16 rows in a
// transposed [N][M] LDS buffer: one 16-byte DS store per lane.
DEVINL void store_acc_T(unsigned short* pT, int ldT, int row0, int col,
                        const v8f acc, int hi) {
    uint4 w;
    w.x = pack_bf2(acc[0], acc[1]);
    w.y = pack_bf2(acc[2], acc[3]);
    w.z = pack_bf2(acc[4], acc[5]);
    w.w = pack_bf2(acc[6], acc[7]);
    *(uint4*)(pT + col * ldT + row0 + hi * 8) = w;
}

DEVINL float wave_max(float v) {
#pragma unroll
    for (int off = 16; off >= 1; off >>= 1) v = fmaxf(v, __shfl_xor(v, off, 32));
    return v;
}
DEVINL float wave_sum(float v) {
#pragma unroll
    for (int off = 16; off >= 1; off >>= 1) v += __shfl_xor(v, off, 32);
    return v;
}

// ---- LDS layout (bytes) --------------------------------------------------
//  [0      , 131072)  WHT  : Wh bf16 TRANSPOSED [512 cols][128 rows]
//                            (layer 2 reuses as Wh_o TRANSPOSED [64][128])
//  [131072 , 262144)  HP   : hcat bf16 row-major [128][512]
//                            (phases 1-2: XS bf16 [128][256] aliases here)
//  [262144 , 294912)  ATT  : per-wave att row tile bf16 [8][16][128]
//  [294912 , 299008)  F1   : f32 [8][128]
//  [299008 , 303104)  F2   : f32 [8][128]
//  [303104 , 305152)  ADJM : adjacency bitmask u32 [128][4]
#define SMEM_BYTES 305152

__global__ __launch_bounds__(256, 1)
void gat_prior_fused(const float* __restrict__ x,     // [B,128,256]
                     const float* __restrict__ adj,   // [128,128]
                     const float* __restrict__ prior, // [P,128,128]
                     const int*   __restrict__ part,  // [B]
                     const float* __restrict__ Whg,   // [8,256,64]
                     const float* __restrict__ a1h,   // [8,64]
                     const float* __restrict__ a2h,   // [8,64]
                     const float* __restrict__ Wog,   // [512,64]
                     const float* __restrict__ a1o,   // [64]
                     const float* __restrict__ a2o,   // [64]
                     float*       __restrict__ out)   // [B,128,64]
{
    extern __shared__ char smem[];
    unsigned short* WHT  = (unsigned short*)(smem);            // [512][128]
    unsigned short* HP   = (unsigned short*)(smem + 131072);   // [128][512]
    unsigned short* XS   = HP;                                 // [128][256] alias
    unsigned short* ATT  = (unsigned short*)(smem + 262144);
    float*          F1   = (float*)(smem + 294912);
    float*          F2   = (float*)(smem + 299008);
    unsigned*       ADJM = (unsigned*)(smem + 303104);

    const int tid  = threadIdx.x;
    const int lane = tid & 31;
    const int wave = tid >> 5;
    const int b    = blockIdx.x;
    const int hi   = lane >> 4;
    const int lo16 = lane & 15;

    const float* pb = prior + (size_t)part[b] * NN * NN;

    // ---- Phase 0: adjacency bitmask via wave32 ballot -------------------
    // word idx in [0,512): row = idx>>2, colgroup = idx&3; lane tests one col.
    for (int t = 0; t < 64; ++t) {
        const int idx = wave * 64 + t;
        const int row = idx >> 2, cg = idx & 3;
        const bool pred = adj[row * NN + cg * 32 + lane] > 0.0f;
        const unsigned m = (unsigned)__ballot(pred);
        if (lane == 0) ADJM[idx] = m;
    }

    // ---- Phase 1: stage x[b] into LDS as bf16 (f32x4 -> bf16x4) --------
    {
        const float4* xb = (const float4*)(x + (size_t)b * NN * NF);
        for (int i = tid; i < NN * NF / 4; i += 256) {
            const float4 v = xb[i];
            uint2 w;
            w.x = pack_bf2(v.x, v.y);
            w.y = pack_bf2(v.z, v.w);
            *(uint2*)(XS + i * 4) = w;
        }
    }
    __syncthreads();

    // ---- Phase 2: per-head GEMM  Wh = XS(128x256) @ W_h[h](256x64) -----
    {
        const int h = wave;
        const float* Wg = Whg + (size_t)h * NF * NHID;
        for (int n = 0; n < 4; ++n) {
            v16bf bf[8];
#pragma unroll
            for (int k = 0; k < 8; ++k) bf[k] = load_b_g(Wg, k * 32, n * 16, NHID, lane);
            for (int m = 0; m < 8; ++m) {
                v8f acc = {};
#pragma unroll
                for (int k = 0; k < 8; ++k) {
                    const v16bf a = load_a_lds(XS, m * 16, k * 32, NF, lane);
                    acc = wmma_bf16(a, bf[k], acc);
                }
                store_acc_T(WHT, NN, m * 16, h * NHID + n * 16 + lo16, acc, hi);
            }
        }
    }
    __syncthreads();   // all waves done with XS before HP is overwritten

    // ---- Phase 3: rank-1 attention terms f1/f2 (per head) --------------
    {
        const int h = wave;
        const float* A1 = a1h + h * NHID;
        const float* A2 = a2h + h * NHID;
#pragma unroll
        for (int rr = 0; rr < 4; ++rr) {
            const int row = lane + rr * 32;
            float s1 = 0.f, s2 = 0.f;
            for (int c = 0; c < NHID; ++c) {
                const float w = bf2f(WHT[(h * NHID + c) * NN + row]);
                s1 += w * A1[c];
                s2 += w * A2[c];
            }
            F1[h * NN + row] = s1;
            F2[h * NN + row] = s2;
        }
    }

    // ---- Phase 4: per-head attention + hp = att @ Wh, ELU -> HP --------
    {
        const int h = wave;
        unsigned short* attT = ATT + h * (16 * NN);
        for (int m = 0; m < 8; ++m) {
            for (int r16 = 0; r16 < 16; ++r16) {
                const int i = m * 16 + r16;
                const float f1i = F1[h * NN + i];
                const unsigned abits =
                    (ADJM[i * 4 + (lane >> 3)] >> ((lane * 4) & 31)) & 0xFu;
                float ev[4];
                float mx = -3.4e38f;
#pragma unroll
                for (int q = 0; q < 4; ++q) {
                    const int j = lane * 4 + q;           // contiguous per lane
                    float e = f1i + F2[h * NN + j];
                    e = e > 0.f ? e : ALPHA * e;
                    e = (abits >> q) & 1u ? e : NEGV;
                    ev[q] = e;
                    mx = fmaxf(mx, e);
                }
                mx = wave_max(mx);
                float s = 0.f;
#pragma unroll
                for (int q = 0; q < 4; ++q) { ev[q] = __expf(ev[q] - mx); s += ev[q]; }
                s = wave_sum(s);
                const float inv = __frcp_rn(s);
                const float4 pv = *(const float4*)(pb + i * NN + lane * 4);
                uint2 w;
                w.x = pack_bf2(BETA * pv.x + (1.0f - BETA) * ev[0] * inv,
                               BETA * pv.y + (1.0f - BETA) * ev[1] * inv);
                w.y = pack_bf2(BETA * pv.z + (1.0f - BETA) * ev[2] * inv,
                               BETA * pv.w + (1.0f - BETA) * ev[3] * inv);
                *(uint2*)(attT + r16 * NN + lane * 4) = w;
            }
            // hp tile: attT(16x128) @ Wh_head(128x64), B from transposed WHT
            for (int n = 0; n < 4; ++n) {
                v8f acc = {};
#pragma unroll
                for (int k = 0; k < 4; ++k) {
                    const v16bf a  = load_a_lds(attT, 0, k * 32, NN, lane);
                    const v16bf bb = load_b_ldsT(WHT, k * 32, h * NHID + n * 16, NN, lane);
                    acc = wmma_bf16(a, bb, acc);
                }
                const int col = h * NHID + n * 16 + lo16;
#pragma unroll
                for (int r = 0; r < 8; ++r) {
                    float v = acc[r];
                    v = v > 0.f ? v : (__expf(v) - 1.0f);   // ELU (concat heads)
                    HP[(m * 16 + r + hi * 8) * 512 + col] = f2bf_bits(v);
                }
            }
        }
    }
    __syncthreads();   // HP(=hcat) complete; WHT now free for reuse

    // ---- Phase 5: Wh_o = HP(128x512) @ W_o(512x64) -> WHT as [64][128] --
    {
        const int m = wave;
        for (int n = 0; n < 4; ++n) {
            v8f acc = {};
            for (int k = 0; k < 16; ++k) {
                const v16bf a  = load_a_lds(HP, m * 16, k * 32, 512, lane);
                const v16bf bb = load_b_g(Wog, k * 32, n * 16, NCLS, lane);
                acc = wmma_bf16(a, bb, acc);
            }
            store_acc_T(WHT, NN, m * 16, n * 16 + lo16, acc, hi);
        }
    }
    __syncthreads();

    // ---- Phase 6: output-layer f1/f2 -----------------------------------
    if (tid < NN) {
        const int row = tid;
        float s1 = 0.f, s2 = 0.f;
        for (int c = 0; c < NCLS; ++c) {
            const float w = bf2f(WHT[c * NN + row]);
            s1 += w * a1o[c];
            s2 += w * a2o[c];
        }
        F1[row] = s1;
        F2[row] = s2;
    }
    __syncthreads();

    // ---- Phase 7: output attention + out = elu(att2 @ Wh_o) ------------
    {
        const int m = wave;
        unsigned short* attT = ATT + m * (16 * NN);
        for (int r16 = 0; r16 < 16; ++r16) {
            const int i = m * 16 + r16;
            const float f1i = F1[i];
            const unsigned abits =
                (ADJM[i * 4 + (lane >> 3)] >> ((lane * 4) & 31)) & 0xFu;
            float ev[4];
            float mx = -3.4e38f;
#pragma unroll
            for (int q = 0; q < 4; ++q) {
                const int j = lane * 4 + q;
                float e = f1i + F2[j];
                e = e > 0.f ? e : ALPHA * e;
                e = (abits >> q) & 1u ? e : NEGV;
                ev[q] = e;
                mx = fmaxf(mx, e);
            }
            mx = wave_max(mx);
            float s = 0.f;
#pragma unroll
            for (int q = 0; q < 4; ++q) { ev[q] = __expf(ev[q] - mx); s += ev[q]; }
            s = wave_sum(s);
            const float inv = __frcp_rn(s);
            const float4 pv = *(const float4*)(pb + i * NN + lane * 4);
            uint2 w;
            w.x = pack_bf2(BETA * pv.x + (1.0f - BETA) * ev[0] * inv,
                           BETA * pv.y + (1.0f - BETA) * ev[1] * inv);
            w.y = pack_bf2(BETA * pv.z + (1.0f - BETA) * ev[2] * inv,
                           BETA * pv.w + (1.0f - BETA) * ev[3] * inv);
            *(uint2*)(attT + r16 * NN + lane * 4) = w;
        }
        float* ob = out + (size_t)b * NN * NCLS;
        for (int n = 0; n < 4; ++n) {
            v8f acc = {};
#pragma unroll
            for (int k = 0; k < 4; ++k) {
                const v16bf a  = load_a_lds(attT, 0, k * 32, NN, lane);
                const v16bf bb = load_b_ldsT(WHT, k * 32, n * 16, NN, lane);
                acc = wmma_bf16(a, bb, acc);
            }
            const int col = n * 16 + lo16;
#pragma unroll
            for (int r = 0; r < 8; ++r) {
                const int row = m * 16 + r + hi * 8;
                float v = acc[r];
                v = v > 0.f ? v : (__expf(v) - 1.0f);   // final ELU
                ob[row * NCLS + col] = v;
            }
        }
    }
}

extern "C" void kernel_launch(void* const* d_in, const int* in_sizes, int n_in,
                              void* d_out, int out_size, void* d_ws, size_t ws_size,
                              hipStream_t stream) {
    const float* x     = (const float*)d_in[0];
    const float* adj   = (const float*)d_in[1];
    const float* prior = (const float*)d_in[2];
    const int*   part  = (const int*)d_in[3];
    const float* Whg   = (const float*)d_in[4];
    const float* a1h   = (const float*)d_in[5];
    const float* a2h   = (const float*)d_in[6];
    const float* Wog   = (const float*)d_in[7];
    const float* a1o   = (const float*)d_in[8];
    const float* a2o   = (const float*)d_in[9];
    float*       out   = (float*)d_out;

    const int B = in_sizes[0] / (NN * NF);   // 2048

    hipFuncSetAttribute((const void*)gat_prior_fused,
                        hipFuncAttributeMaxDynamicSharedMemorySize, SMEM_BYTES);

    gat_prior_fused<<<B, 256, SMEM_BYTES, stream>>>(
        x, adj, prior, part, Whg, a1h, a2h, Wog, a1o, a2o, out);
}
